// pretrained_Resnet_LSTM_Decoder_83245056131859
// MI455X (gfx1250) — matrix operand order repaired and checked by reading, not compiled, not run on hardware
//
#include <hip/hip_runtime.h>
#include <math.h>

// ---------------------------------------------------------------------------
// Types for CDNA5 WMMA (wave32, v_wmma_f32_16x16x32_bf16)
// ---------------------------------------------------------------------------
typedef __attribute__((ext_vector_type(16))) __bf16 v16bf;
typedef __attribute__((ext_vector_type(8)))  __bf16 v8bf;
typedef __attribute__((ext_vector_type(8)))  float  v8f;

#define BB 128   // batch
#define TT 17    // timesteps
#define EE 512   // embed
#define HH 512   // hidden
#define GG 2048  // 4*H gates
#define DD 1024  // D2
#define VV 32000 // vocab
#define MROWS (BB*TT) // 2176

// ---------------------------------------------------------------------------
// Generic bf16 WMMA GEMM:  C[M,N] = A[M,K] @ Bt[N,K]^T (+bias +addend, relu)
// One wave -> 32(M) x 64(N) tile: 2 A fragments, 4 B fragments, 8 accums.
// Each B fragment feeds 2 WMMAs -> 8 WMMAs per 12 b128 loads per k-step.
// Block = 128 thr = 4 waves covering 32 x 256 outputs.
// grid = (N/256, M/32). All problem dims here divide exactly.
// ---------------------------------------------------------------------------
__global__ __launch_bounds__(128)
void wmma_gemm_bf16(const __bf16* __restrict__ A,   // M x K row-major
                    const __bf16* __restrict__ Bt,  // N x K row-major (B^T)
                    const float*  __restrict__ bias,    // [N] or null
                    const float*  __restrict__ addend,  // fp32, row stride ld_add, or null
                    int ld_add,
                    float*  __restrict__ outF,          // fp32 out (ldc) or null
                    __bf16* __restrict__ outB,          // bf16 out (ldc) or null
                    int ldc, int N, int K, int relu)
{
    const int wave  = threadIdx.x >> 5;
    const int lane  = threadIdx.x & 31;
    const int hi    = lane >> 4;      // lane 16..31
    const int lo    = lane & 15;
    const int mBase = blockIdx.y * 32;
    const int nBase = (blockIdx.x * 4 + wave) * 64;

    const v8f zero = {0.f,0.f,0.f,0.f,0.f,0.f,0.f,0.f};
    v8f acc[2][4];
    #pragma unroll
    for (int mh = 0; mh < 2; ++mh)
        #pragma unroll
        for (int nt = 0; nt < 4; ++nt) acc[mh][nt] = zero;

    const __bf16* aRow0 = A + (size_t)(mBase + lo) * K;        // rows m..m+15
    const __bf16* aRow1 = A + (size_t)(mBase + 16 + lo) * K;   // rows m+16..m+31

    for (int kb = 0; kb < K; kb += 32) {
        // A fragments (16x32 tiles): 16 contiguous bf16 per lane (k = 16*hi + s)
        v8bf a00 = *(const v8bf*)(aRow0 + kb + hi * 16);
        v8bf a01 = *(const v8bf*)(aRow0 + kb + hi * 16 + 8);
        v16bf a0 = __builtin_shufflevector(a00, a01,
                     0,1,2,3,4,5,6,7,8,9,10,11,12,13,14,15);
        v8bf a10 = *(const v8bf*)(aRow1 + kb + hi * 16);
        v8bf a11 = *(const v8bf*)(aRow1 + kb + hi * 16 + 8);
        v16bf a1 = __builtin_shufflevector(a10, a11,
                     0,1,2,3,4,5,6,7,8,9,10,11,12,13,14,15);
        // B fragments (32x16 tiles): 16 contiguous bf16 per lane from Bt
        const __bf16* bBase = Bt + (size_t)(nBase + lo) * K + kb + hi * 16;
        #pragma unroll
        for (int nt = 0; nt < 4; ++nt) {
            v16bf b = *(const v16bf*)(bBase + (size_t)(nt * 16) * K);
            acc[0][nt] = __builtin_amdgcn_wmma_f32_16x16x32_bf16(
                false, a0, false, b, (short)0, acc[0][nt], false, false);
            acc[1][nt] = __builtin_amdgcn_wmma_f32_16x16x32_bf16(
                false, a1, false, b, (short)0, acc[1][nt], false, false);
        }
    }

    #pragma unroll
    for (int nt = 0; nt < 4; ++nt) {
        const int n  = nBase + nt * 16 + lo;
        const float bv = bias ? bias[n] : 0.f;
        #pragma unroll
        for (int mh = 0; mh < 2; ++mh) {
            #pragma unroll
            for (int r = 0; r < 8; ++r) {
                const int m = mBase + mh * 16 + hi * 8 + r; // C/D: row = vgpr + 8*hi
                float v = acc[mh][nt][r] + bv;
                if (addend) v += addend[(size_t)m * ld_add + n];
                if (relu)   v = v > 0.f ? v : 0.f;
                if (outF) outF[(size_t)m * ldc + n] = v;
                if (outB) outB[(size_t)m * ldc + n] = (__bf16)v;
            }
        }
    }
}

// ---------------------------------------------------------------------------
// Helpers: convert / transpose / gather / bias / init
// ---------------------------------------------------------------------------
__global__ void convert_f32_bf16(const float* __restrict__ in,
                                 __bf16* __restrict__ out, int n)
{
    int i = blockIdx.x * blockDim.x + threadIdx.x;
    if (i < n) out[i] = (__bf16)in[i];
}

// in: K x N fp32 -> out: N x K bf16 (tiled, LDS, all dims multiples of 32)
__global__ __launch_bounds__(256)
void transpose_f32_bf16(const float* __restrict__ in,
                        __bf16* __restrict__ out, int K, int N)
{
    __shared__ float tile[32][33];
    const int k0 = blockIdx.y * 32, n0 = blockIdx.x * 32;
    for (int r = threadIdx.y; r < 32; r += 8)
        tile[r][threadIdx.x] = in[(size_t)(k0 + r) * N + n0 + threadIdx.x];
    __syncthreads();
    for (int r = threadIdx.y; r < 32; r += 8)
        out[(size_t)(n0 + r) * K + k0 + threadIdx.x] =
            (__bf16)tile[threadIdx.x][r];
}

__global__ void gather_emb(const float* __restrict__ emb,
                           const int* __restrict__ cap,
                           __bf16* __restrict__ Xbf)
{
    int idx = blockIdx.x * blockDim.x + threadIdx.x;   // B*16*E
    if (idx >= BB * 16 * EE) return;
    const int j  = idx & (EE - 1);
    const int bt = idx >> 9;
    const int b  = bt >> 4, tt = bt & 15;
    const int tok = cap[b * TT + tt];
    Xbf[((size_t)b * TT + tt + 1) * EE + j] = (__bf16)emb[(size_t)tok * EE + j];
}

__global__ void bias_combine(const float* __restrict__ b_ih,
                             const float* __restrict__ b_hh,
                             float* __restrict__ out)
{
    int i = blockIdx.x * blockDim.x + threadIdx.x;
    if (i < GG) out[i] = b_ih[i] + b_hh[i];
}

__global__ void init_state(__bf16* __restrict__ h_bf, float* __restrict__ c)
{
    int i = blockIdx.x * blockDim.x + threadIdx.x;
    if (i < BB * HH) { h_bf[i] = (__bf16)0.f; c[i] = 0.f; }
}

// ---------------------------------------------------------------------------
// LSTM pointwise cell: gates(128x2048) -> c, h  (gate order i,f,g,o)
// ---------------------------------------------------------------------------
__global__ void lstm_cell(const float* __restrict__ gates,
                          float* __restrict__ c,
                          __bf16* __restrict__ h_bf,
                          __bf16* __restrict__ hs_bf, int t)
{
    int idx = blockIdx.x * blockDim.x + threadIdx.x;   // B*H
    if (idx >= BB * HH) return;
    const int b = idx >> 9, j = idx & (HH - 1);
    const float* g = gates + (size_t)b * GG;
    const float ig = 1.f / (1.f + __expf(-g[j]));
    const float fg = 1.f / (1.f + __expf(-g[HH + j]));
    const float gg = tanhf(g[2 * HH + j]);
    const float og = 1.f / (1.f + __expf(-g[3 * HH + j]));
    const float cv = fg * c[idx] + ig * gg;
    c[idx] = cv;
    const float h = og * tanhf(cv);
    h_bf[idx] = (__bf16)h;
    hs_bf[((size_t)b * TT + t) * HH + j] = (__bf16)h;
}

// ---------------------------------------------------------------------------
// In-place row softmax over V=32000 (one block / row)
// ---------------------------------------------------------------------------
__global__ __launch_bounds__(256)
void softmax_rows(float* __restrict__ logits, int N)
{
    __shared__ float red[256];
    float* row = logits + (size_t)blockIdx.x * N;
    const int tid = threadIdx.x;

    float m = -INFINITY;
    for (int i = tid; i < N; i += 256) m = fmaxf(m, row[i]);
    red[tid] = m; __syncthreads();
    for (int s = 128; s > 0; s >>= 1) {
        if (tid < s) red[tid] = fmaxf(red[tid], red[tid + s]);
        __syncthreads();
    }
    m = red[0]; __syncthreads();

    float sum = 0.f;
    for (int i = tid; i < N; i += 256) {
        float e = __expf(row[i] - m);
        row[i] = e;
        sum += e;
    }
    red[tid] = sum; __syncthreads();
    for (int s = 128; s > 0; s >>= 1) {
        if (tid < s) red[tid] += red[tid + s];
        __syncthreads();
    }
    const float inv = 1.f / red[0];
    for (int i = tid; i < N; i += 256) row[i] *= inv;
}

// ---------------------------------------------------------------------------
// Host orchestration
// ---------------------------------------------------------------------------
extern "C" void kernel_launch(void* const* d_in, const int* in_sizes, int n_in,
                              void* d_out, int out_size, void* d_ws, size_t ws_size,
                              hipStream_t stream)
{
    const float* features = (const float*)d_in[0];
    const int*   captions = (const int*)  d_in[1];
    const float* W_enc    = (const float*)d_in[2];
    const float* b_enc    = (const float*)d_in[3];
    const float* emb      = (const float*)d_in[4];
    const float* W_ih     = (const float*)d_in[5];
    const float* b_ih     = (const float*)d_in[6];
    const float* W_hh     = (const float*)d_in[7];
    const float* b_hh     = (const float*)d_in[8];
    const float* W_d2     = (const float*)d_in[9];
    const float* b_d2     = (const float*)d_in[10];
    const float* W_last   = (const float*)d_in[11];
    const float* b_last   = (const float*)d_in[12];
    float* out = (float*)d_out;

    // workspace carve-out (256B aligned), ~95 MB total
    char* p = (char*)d_ws;
    auto alloc = [&](size_t bytes) -> void* {
        void* r = (void*)p;
        p += (bytes + 255) & ~(size_t)255;
        return r;
    };
    __bf16* W_ih_bf  = (__bf16*)alloc((size_t)GG * EE * 2);      // (2048,512) already (N,K)
    __bf16* W_hh_bf  = (__bf16*)alloc((size_t)GG * HH * 2);      // (2048,512) already (N,K)
    __bf16* W_encT   = (__bf16*)alloc((size_t)EE * EE * 2);      // (512,512) -> (N,K)
    __bf16* W_d2T    = (__bf16*)alloc((size_t)DD * HH * 2);      // (512,1024)->(1024,512)
    __bf16* W_lastT  = (__bf16*)alloc((size_t)VV * DD * 2);      // (1024,32000)->(32000,1024)
    __bf16* feat_bf  = (__bf16*)alloc((size_t)BB * EE * 2);
    __bf16* Xbf      = (__bf16*)alloc((size_t)MROWS * EE * 2);   // (B,17,E)
    float*  xW       = (float*) alloc((size_t)MROWS * GG * 4);   // (B,17,4H)
    float*  biasc    = (float*) alloc((size_t)GG * 4);
    __bf16* h_bf     = (__bf16*)alloc((size_t)BB * HH * 2);
    float*  c_st     = (float*) alloc((size_t)BB * HH * 4);
    float*  gates    = (float*) alloc((size_t)BB * GG * 4);
    __bf16* hs_bf    = (__bf16*)alloc((size_t)MROWS * HH * 2);
    __bf16* out_bf   = (__bf16*)alloc((size_t)MROWS * DD * 2);
    (void)ws_size; (void)n_in; (void)in_sizes; (void)out_size;

    const dim3 blk1(256);
    // --- weight preparation ---
    convert_f32_bf16<<<dim3((GG*EE + 255)/256), blk1, 0, stream>>>(W_ih, W_ih_bf, GG*EE);
    convert_f32_bf16<<<dim3((GG*HH + 255)/256), blk1, 0, stream>>>(W_hh, W_hh_bf, GG*HH);
    transpose_f32_bf16<<<dim3(EE/32, EE/32), dim3(32,8), 0, stream>>>(W_enc, W_encT, EE, EE);
    transpose_f32_bf16<<<dim3(DD/32, HH/32), dim3(32,8), 0, stream>>>(W_d2, W_d2T, HH, DD);
    transpose_f32_bf16<<<dim3(VV/32, DD/32), dim3(32,8), 0, stream>>>(W_last, W_lastT, DD, VV);
    convert_f32_bf16<<<dim3((BB*EE + 255)/256), blk1, 0, stream>>>(features, feat_bf, BB*EE);
    bias_combine<<<dim3((GG + 255)/256), blk1, 0, stream>>>(b_ih, b_hh, biasc);
    init_state<<<dim3((BB*HH + 255)/256), blk1, 0, stream>>>(h_bf, c_st);

    // --- encoder GEMM: feat = features @ W_enc + b_enc  -> Xbf[:,0,:] ---
    wmma_gemm_bf16<<<dim3(EE/256, BB/32), dim3(128), 0, stream>>>(
        feat_bf, W_encT, b_enc, nullptr, 0,
        nullptr, Xbf, TT*EE, EE, EE, 0);

    // --- embedding gather -> Xbf[:,1:,:] ---
    gather_emb<<<dim3((BB*16*EE + 255)/256), blk1, 0, stream>>>(emb, captions, Xbf);

    // --- precompute xW = X @ W_ih^T + (b_ih + b_hh), all timesteps batched ---
    wmma_gemm_bf16<<<dim3(GG/256, MROWS/32), dim3(128), 0, stream>>>(
        Xbf, W_ih_bf, biasc, nullptr, 0,
        xW, nullptr, GG, GG, EE, 0);

    // --- LSTM recurrence (sequential in t) ---
    for (int t = 0; t < TT; ++t) {
        wmma_gemm_bf16<<<dim3(GG/256, BB/32), dim3(128), 0, stream>>>(
            h_bf, W_hh_bf, nullptr, xW + (size_t)t * GG, TT*GG,
            gates, nullptr, GG, GG, HH, 0);
        lstm_cell<<<dim3((BB*HH + 255)/256), blk1, 0, stream>>>(
            gates, c_st, h_bf, hs_bf, t);
    }

    // --- out = relu(hs @ W_d2 + b_d2) ---
    wmma_gemm_bf16<<<dim3(DD/256, MROWS/32), dim3(128), 0, stream>>>(
        hs_bf, W_d2T, b_d2, nullptr, 0,
        nullptr, out_bf, DD, DD, HH, 1);

    // --- logits = out @ W_last + b_last  (dominant GEMM, fp32 -> d_out) ---
    wmma_gemm_bf16<<<dim3(VV/256, MROWS/32), dim3(128), 0, stream>>>(
        out_bf, W_lastT, b_last, nullptr, 0,
        out, nullptr, VV, VV, DD, 0);

    // --- softmax over vocab, in place on d_out ---
    softmax_rows<<<dim3(MROWS), dim3(256), 0, stream>>>(out, VV);
}
